// DeepEMD_10668698763879
// MI455X (gfx1250) — compile-verified, matrix operand order checked by reading, not run on hardware
//
#include <hip/hip_runtime.h>

typedef __attribute__((ext_vector_type(16))) _Float16 v16h;
typedef __attribute__((ext_vector_type(8)))  float    v8f;

#define B_    8
#define NQ_   75
#define WAY_  5
#define SHOT_ 5
#define D_    64
#define N_    100
#define NPAD  112      // 7 tiles of 16
#define NT    7
#define KST   113      // LDS row stride in floats; gcd(113,64)=1 -> conflict-free column sweeps
#define REG_  0.1f
#define EPS_  1e-8f
#define ITERS 10

// ---------------------------------------------------------------------------
// Kernel 1a: normalize q over D, emit f16 transposed rows qt[bq][n][d]
// (rows n>=100 zeroed so WMMA padding contributes exact zeros)
// ---------------------------------------------------------------------------
__global__ void qt_prep(const float* __restrict__ q, _Float16* __restrict__ qt) {
    const int bq = blockIdx.x;
    const int t  = threadIdx.x;           // column n
    if (t >= NPAD) return;
    const float* qb = q + (size_t)bq * (D_ * N_);
    _Float16* row = qt + (size_t)bq * NPAD * D_ + (size_t)t * D_;
    if (t < N_) {
        float ss = 0.f;
        #pragma unroll 8
        for (int d = 0; d < D_; ++d) { float x = qb[d * N_ + t]; ss += x * x; }
        float inv = 1.0f / (sqrtf(ss) + 1e-12f);
        #pragma unroll 8
        for (int d = 0; d < D_; ++d) row[d] = (_Float16)(qb[d * N_ + t] * inv);
    } else {
        #pragma unroll 8
        for (int d = 0; d < D_; ++d) row[d] = (_Float16)0.0f;
    }
}

// ---------------------------------------------------------------------------
// Kernel 1b: shot-normalize + shot-mean of s, emit f16 transposed st[bw][n][d]
// ---------------------------------------------------------------------------
__global__ void st_prep(const float* __restrict__ s, _Float16* __restrict__ st) {
    const int bw = blockIdx.x;            // b*WAY_ + w
    const int t  = threadIdx.x;           // column n
    if (t >= NPAD) return;
    const float* sb = s + (size_t)bw * SHOT_ * D_ * N_;
    _Float16* row = st + (size_t)bw * NPAD * D_ + (size_t)t * D_;
    if (t < N_) {
        for (int d = 0; d < D_; ++d) {
            float sum = 0.f, ss = 0.f;
            #pragma unroll
            for (int sh = 0; sh < SHOT_; ++sh) {
                float x = sb[(size_t)sh * D_ * N_ + d * N_ + t];
                sum += x; ss += x * x;
            }
            row[d] = (_Float16)(sum / ((float)SHOT_ * (sqrtf(ss) + 1e-12f)));
        }
    } else {
        #pragma unroll 8
        for (int d = 0; d < D_; ++d) row[d] = (_Float16)0.0f;
    }
}

// ---------------------------------------------------------------------------
// Kernel 2: one workgroup (4 wave32) per (bq, w): WMMA Gram -> K in LDS ->
// Sinkhorn -> score. Fragment layout per ISA 7.12.2 (16-bit A 16x32): lane
// holds row M=lane%16; k-pairs split by half-wave as two contiguous 16B runs.
// ---------------------------------------------------------------------------
__global__ void emd_kernel(const _Float16* __restrict__ qt,
                           const _Float16* __restrict__ st,
                           float* __restrict__ out) {
    extern __shared__ char smem[];
    float* Kmat = (float*)smem;                 // NPAD * KST floats
    float* uu   = Kmat + NPAD * KST;            // 128
    float* vv   = uu + 128;                     // 128
    float* red  = vv + 128;                     // 128

    const int p    = blockIdx.x;
    const int bq   = p / WAY_;
    const int w    = p % WAY_;
    const int b    = bq / NQ_;
    const int t    = threadIdx.x;
    const int lane = t & 31;
    const int wave = t >> 5;
    const int half = (lane >> 4) & 1;
    const int lrow = lane & 15;

    const uint4* qbase = (const uint4*)(qt + (size_t)bq * NPAD * D_);
    const uint4* sbase = (const uint4*)(st + ((size_t)b * WAY_ + w) * NPAD * D_);
    // each row = 64 halves = 8 uint4

    // ---- Gram matrix: 49 tiles split across 4 waves (uniform per wave) ----
    for (int tile = wave; tile < NT * NT; tile += 4) {
        const int ti = tile / NT, tj = tile % NT;
        const uint4* arow = qbase + (size_t)(ti * 16 + lrow) * 8;
        const uint4* brow = sbase + (size_t)(tj * 16 + lrow) * 8;
        v8f c = {};
        #pragma unroll
        for (int kb = 0; kb < 2; ++kb) {
            union { struct { uint4 lo, hi; } u; v16h h; } ua, ub;
            ua.u.lo = arow[kb * 4 + half];
            ua.u.hi = arow[kb * 4 + 2 + half];
            ub.u.lo = brow[kb * 4 + half];
            ub.u.hi = brow[kb * 4 + 2 + half];
            c = __builtin_amdgcn_wmma_f32_16x16x32_f16(
                    /*neg_a=*/false, ua.h, /*neg_b=*/false, ub.h,
                    /*c_mod=*/(short)0, c, /*reuse_a=*/false, /*reuse_b=*/false);
        }
        const int m  = tj * 16 + lrow;
        const int nb = ti * 16 + half * 8;
        #pragma unroll
        for (int r = 0; r < 8; ++r) {
            const int n = nb + r;
            const float cost = 1.0f - c[r];
            const float kv = (n < N_ && m < N_) ? __expf(-cost * (1.0f / REG_)) : 0.0f;
            Kmat[n * KST + m] = kv;
        }
    }

    uu[t] = (t < N_) ? 1.0f : 0.0f;
    vv[t] = 0.0f;
    __syncthreads();

    // ---- Sinkhorn ----
    const float mu = 1.0f / N_, nu = 1.0f / N_;
    for (int it = 0; it < ITERS; ++it) {
        if (t < N_) {                      // v_m = nu / (K^T u + eps)
            float s = 0.f;
            for (int n = 0; n < N_; ++n) s += Kmat[n * KST + t] * uu[n];
            vv[t] = nu / (s + EPS_);
        }
        __syncthreads();
        if (t < N_) {                      // u_n = mu / (K v + eps)
            float s = 0.f;
            const float* krow = Kmat + t * KST;
            for (int m = 0; m < N_; ++m) s += krow[m] * vv[m];
            uu[t] = mu / (s + EPS_);
        }
        __syncthreads();
    }

    // ---- dist = sum u_n K_nm v_m cost_nm, cost = -reg*log(K) ----
    float partial = 0.f;
    if (t < N_) {
        const float* krow = Kmat + t * KST;
        const float un = uu[t];
        for (int m = 0; m < N_; ++m) {
            const float kk = krow[m];
            partial += un * kk * vv[m] * (-REG_ * __logf(kk));
        }
    }
    red[t] = partial;
    __syncthreads();
    #pragma unroll
    for (int s_ = 64; s_ > 0; s_ >>= 1) {
        if (t < s_) red[t] += red[t + s_];
        __syncthreads();
    }
    if (t == 0) out[p] = -red[0];
}

// ---------------------------------------------------------------------------
extern "C" void kernel_launch(void* const* d_in, const int* in_sizes, int n_in,
                              void* d_out, int out_size, void* d_ws, size_t ws_size,
                              hipStream_t stream) {
    const float* q = (const float*)d_in[0];   // (600, 64, 100) f32
    const float* s = (const float*)d_in[1];   // (8, 5, 5, 64, 100) f32
    float* out = (float*)d_out;               // (600, 5) f32

    _Float16* qt = (_Float16*)d_ws;                              // 600*112*64 halves
    _Float16* st = qt + (size_t)(B_ * NQ_) * NPAD * D_;          // 40*112*64 halves

    qt_prep<<<B_ * NQ_, 128, 0, stream>>>(q, qt);
    st_prep<<<B_ * WAY_, 128, 0, stream>>>(s, st);

    const size_t smem = (size_t)(NPAD * KST + 3 * 128) * sizeof(float); // ~51 KB
    emd_kernel<<<(B_ * NQ_) * WAY_, 128, smem, stream>>>(qt, st, out);
}
